// HConstructorfor_graph_7121055777190
// MI455X (gfx1250) — matrix operand order repaired and checked by reading, not compiled, not run on hardware
//
#include <hip/hip_runtime.h>
#include <hip/hip_bf16.h>

typedef __attribute__((ext_vector_type(16))) _Float16 v16h;
typedef __attribute__((ext_vector_type(8)))  float    v8f;

#define N_NODES 20000
#define N_EDGES 160000
#define T_REP   3
#define DIM     128
#define NT      80000                         // (T+1)*N
#define TOT_E   (2*N_EDGES + 3*N_NODES)       // 380000 scatter edges (loops handled in init)
#define HCHUNK  2000

static __device__ __forceinline__ float warp_sum(float v) {
#pragma unroll
  for (int off = 16; off > 0; off >>= 1) v += __shfl_xor(v, off, 32);
  return v;
}

// ---------------------------------------------------------------------------
// Generic 128-wide WMMA GEMM: C[M,128] = alpha * A[M,128] @ W + bias
// A: f16 row-major. W: f32, [K=128][N=128] row-major (or [N][K] if W_T).
// Per block: 256 threads = 8 waves, 128 rows. Wave = 16-row strip, 8 N-tiles.
// ---------------------------------------------------------------------------
template<bool RELU_A, bool W_T>
__global__ __launch_bounds__(256)
void gemm128_wmma(const _Float16* __restrict__ A, const float* __restrict__ W,
                  const float* __restrict__ bias, float* __restrict__ Cf,
                  _Float16* __restrict__ Ch, int M, float alpha)
{
  __shared__ alignas(32) _Float16 ldsW[DIM * DIM];   // Wt[n][k], 32 KB
  const int tid  = threadIdx.x;
  const int wave = tid >> 5;
  const int lane = tid & 31;
  const int hi   = lane >> 4;
  const int lq   = lane & 15;

  // stage weights to LDS as f16, transposed to [n][k]
  for (int i = tid; i < DIM * DIM; i += 256) {
    if (W_T) {
      ldsW[i] = (_Float16)W[i];                      // already [n][k]
    } else {
      int k = i >> 7, n = i & 127;
      ldsW[n * DIM + k] = (_Float16)W[i];
    }
  }
  __syncthreads();

  const int rowBase = blockIdx.x * 128 + wave * 16;
  int arow = rowBase + lq; if (arow >= M) arow = M - 1;   // clamp (safe read)
  const _Float16* ap = A + (size_t)arow * DIM;

  v8f zero = {};
  v8f acc[8];
#pragma unroll
  for (int i = 0; i < 8; ++i) acc[i] = zero;

#pragma unroll
  for (int ks = 0; ks < 4; ++ks) {                   // K step of 32
    const int kb = ks * 32;
    union { v16h v; _Float16 e[16]; } af;
#pragma unroll
    for (int q = 0; q < 8; ++q) {                    // contiguous k = 16*hi + s (matches B)
      int k0 = kb + hi * 16 + q * 2;
      float x0 = (float)ap[k0];
      float x1 = (float)ap[k0 + 1];
      if (RELU_A) { x0 = fmaxf(x0, 0.f); x1 = fmaxf(x1, 0.f); }
      af.e[2 * q]     = (_Float16)x0;
      af.e[2 * q + 1] = (_Float16)x1;
    }
#pragma unroll
    for (int nt = 0; nt < 8; ++nt) {
      int ncol = nt * 16 + lq;
      v16h bf = *(const v16h*)(ldsW + ncol * DIM + kb + hi * 16);
      acc[nt] = __builtin_amdgcn_wmma_f32_16x16x32_f16(
          false, af.v, false, bf, (short)0, acc[nt], false, false);
    }
  }

#pragma unroll
  for (int nt = 0; nt < 8; ++nt) {
    int n = nt * 16 + lq;
    float b = bias ? bias[n] : 0.f;
    union { v8f v; float e[8]; } u; u.v = acc[nt];
#pragma unroll
    for (int r = 0; r < 8; ++r) {
      int m = rowBase + hi * 8 + r;                  // ISA 32-bit C/D layout
      if (m < M) {
        float val = u.e[r] * alpha + b;
        Cf[(size_t)m * DIM + n] = val;
        if (Ch) Ch[(size_t)m * DIM + n] = (_Float16)val;
      }
    }
  }
}

// ---------------------------------------------------------------------------
__global__ void prep_features(const float* __restrict__ f, float* __restrict__ af,
                              _Float16* __restrict__ ah, int n) {
  int i = blockIdx.x * blockDim.x + threadIdx.x;
  if (i < n) { float v = f[i]; af[i] = v; ah[i] = (_Float16)v; }
}

__global__ void fill_f32(float* __restrict__ p, float v, int n) {
  int i = blockIdx.x * blockDim.x + threadIdx.x;
  if (i < n) p[i] = v;
}

__global__ void row_norms(const float* __restrict__ x, float* __restrict__ norms, int rows) {
  int w = (blockIdx.x * blockDim.x + threadIdx.x) >> 5;
  int lane = threadIdx.x & 31;
  if (w >= rows) return;
  const float* rp = x + (size_t)w * DIM;
  float s = 0.f;
#pragma unroll
  for (int j = 0; j < 4; ++j) { float v = rp[lane + 32 * j]; s += v * v; }
  s = warp_sum(s);
  if (lane == 0) norms[w] = sqrtf(s);
}

// cosine-sim replica argmax, faithful to the reference's repeat/tile quirk:
// sim[e,i] = cos(features[src[e]], transformed[dst[(e*T+i)%E]])
__global__ void edge_argmax(const int* __restrict__ ei, const float* __restrict__ af,
                            const float* __restrict__ norms, int* __restrict__ best_nodes) {
  int e = (blockIdx.x * blockDim.x + threadIdx.x) >> 5;
  int lane = threadIdx.x & 31;
  if (e >= N_EDGES) return;
  const int* dstp = ei + N_EDGES;
  int s = ei[e];
  const float* apn = af + (size_t)s * DIM;
  float a0 = apn[lane], a1 = apn[lane + 32], a2 = apn[lane + 64], a3 = apn[lane + 96];
  float na = norms[s];
  float bestv = -1e30f; int besti = 0;
#pragma unroll
  for (int i = 0; i < T_REP; ++i) {
    int nb = dstp[(e * T_REP + i) % N_EDGES];
    const float* bp = af + (size_t)(N_NODES + nb) * DIM;   // transformed row (replica block 0 base)
    float p = a0 * bp[lane] + a1 * bp[lane + 32] + a2 * bp[lane + 64] + a3 * bp[lane + 96];
    p = warp_sum(p);
    float sim = p / fmaxf(na * norms[N_NODES + nb], 1e-8f);
    if (sim > bestv) { bestv = sim; besti = i; }           // first-max tie rule
  }
  if (lane == 0) best_nodes[e] = dstp[e] + besti * N_NODES;
}

__global__ void deg_count(const int* __restrict__ ei, const int* __restrict__ best_nodes,
                          float* __restrict__ deg) {
  int i = blockIdx.x * blockDim.x + threadIdx.x;
  if (i < N_EDGES) {
    atomicAdd(&deg[ei[N_EDGES + i]], 1.f);     // dst
    atomicAdd(&deg[best_nodes[i]], 1.f);       // best replica target
  } else {
    int j = i - N_EDGES;
    if (j < 3 * N_NODES) {
      int v = j % N_NODES, rep = j / N_NODES;  // extra_dst = v + rep*N
      atomicAdd(&deg[v + rep * N_NODES], 1.f);
    }
  }
}

__global__ void make_dinv(float* __restrict__ deg, int n) {
  int i = blockIdx.x * blockDim.x + threadIdx.x;
  if (i < n) deg[i] = rsqrtf(deg[i]);
}

// out[v][c] = bias[c] + xw[v][c] * dinv[v]^2   (self-loop term)
__global__ void gcn_init(const float* __restrict__ xw, const float* __restrict__ bias,
                         const float* __restrict__ dinv, float* __restrict__ out, int total) {
  int i = blockIdx.x * blockDim.x + threadIdx.x;
  if (i >= total) return;
  int v = i >> 7, c = i & 127;
  float dv = dinv[v];
  out[i] = bias[c] + xw[i] * dv * dv;
}

// edge scatter-add: one wave per edge, 4 atomic f32 adds per lane
__global__ void gcn_scatter(const int* __restrict__ ei, const int* __restrict__ best_nodes,
                            const float* __restrict__ dinv, const float* __restrict__ xw,
                            float* __restrict__ out) {
  int idx = (blockIdx.x * blockDim.x + threadIdx.x) >> 5;
  int lane = threadIdx.x & 31;
  if (idx >= TOT_E) return;
  int s, d;
  if (idx < N_EDGES)            { s = ei[idx];            d = ei[N_EDGES + idx]; }
  else if (idx < 2 * N_EDGES)   { int e = idx - N_EDGES;  s = ei[e]; d = best_nodes[e]; }
  else                          { int j = idx - 2 * N_EDGES; s = j % N_NODES; d = s + (j / N_NODES) * N_NODES; }
  float w = dinv[s] * dinv[d];
  const float* xp = xw + (size_t)s * DIM;
  float* op = out + (size_t)d * DIM;
#pragma unroll
  for (int j = 0; j < 4; ++j) {
    int c = lane + 32 * j;
    atomicAdd(&op[c], xp[c] * w);
  }
}

__global__ void relu_half(const float* __restrict__ in, _Float16* __restrict__ out, int n) {
  int i = blockIdx.x * blockDim.x + threadIdx.x;
  if (i < n) out[i] = (_Float16)fmaxf(in[i], 0.f);
}

__global__ void row_argmax(const float* __restrict__ x, int* __restrict__ cls, int rows) {
  int r = (blockIdx.x * blockDim.x + threadIdx.x) >> 5;
  int lane = threadIdx.x & 31;
  if (r >= rows) return;
  const float* rp = x + (size_t)r * DIM;
  float bv = -1e30f; int bi = 0;
#pragma unroll
  for (int j = 0; j < 4; ++j) {
    int c = lane * 4 + j;
    float v = rp[c];
    if (v > bv) { bv = v; bi = c; }
  }
#pragma unroll
  for (int off = 16; off > 0; off >>= 1) {
    float ov = __shfl_xor(bv, off, 32);
    int   oi = __shfl_xor(bi, off, 32);
    if (ov > bv || (ov == bv && oi < bi)) { bv = ov; bi = oi; }  // first-max
  }
  if (lane == 0) cls[r] = bi;
}

__global__ void vote_count(const int* __restrict__ cls, float* __restrict__ Hcnt, int rows) {
  int r = blockIdx.x * blockDim.x + threadIdx.x;
  if (r >= rows) return;
  atomicAdd(&Hcnt[(size_t)(r % N_NODES) * DIM + cls[r]], 1.f);
}

// hyper[c][d] = sum_{v<N : Hcnt[v][c]>0} feat2[v][d]
__global__ __launch_bounds__(256)
void hyperedge_kernel(const float* __restrict__ Hcnt, const float* __restrict__ feat2,
                      float* __restrict__ hyper) {
  __shared__ float sf[8][DIM];
  __shared__ float sm[8][DIM];
  float acc[64];
#pragma unroll
  for (int i = 0; i < 64; ++i) acc[i] = 0.f;
  int c  = threadIdx.x >> 1;
  int d0 = (threadIdx.x & 1) * 64;
  int v0 = blockIdx.x * HCHUNK;
  int v1 = v0 + HCHUNK; if (v1 > N_NODES) v1 = N_NODES;
  for (int vb = v0; vb < v1; vb += 8) {
    int nrows = v1 - vb; if (nrows > 8) nrows = 8;
    __syncthreads();
    for (int i = threadIdx.x; i < nrows * DIM; i += 256) {
      int vv = i >> 7, d = i & 127;
      sf[vv][d] = feat2[(size_t)(vb + vv) * DIM + d];
      sm[vv][d] = Hcnt[(size_t)(vb + vv) * DIM + d];
    }
    __syncthreads();
    for (int vv = 0; vv < nrows; ++vv) {
      if (sm[vv][c] > 0.f) {
#pragma unroll
        for (int d = 0; d < 64; ++d) acc[d] += sf[vv][d0 + d];
      }
    }
  }
#pragma unroll
  for (int d = 0; d < 64; ++d) atomicAdd(&hyper[(size_t)c * DIM + d0 + d], acc[d]);
}

// softmax over axis 0 (nodes), one block per class column
__global__ __launch_bounds__(256)
void softmax_col(const float* __restrict__ Hcnt, float* __restrict__ Hout) {
  __shared__ float red[256];
  int c = blockIdx.x, tid = threadIdx.x;
  float m = -1e30f;
  for (int v = tid; v < N_NODES; v += 256) m = fmaxf(m, Hcnt[(size_t)v * DIM + c]);
  red[tid] = m; __syncthreads();
  for (int s = 128; s > 0; s >>= 1) { if (tid < s) red[tid] = fmaxf(red[tid], red[tid + s]); __syncthreads(); }
  m = red[0]; __syncthreads();
  float sum = 0.f;
  for (int v = tid; v < N_NODES; v += 256) sum += __expf(Hcnt[(size_t)v * DIM + c] - m);
  red[tid] = sum; __syncthreads();
  for (int s = 128; s > 0; s >>= 1) { if (tid < s) red[tid] += red[tid + s]; __syncthreads(); }
  float inv = 1.f / red[0];
  for (int v = tid; v < N_NODES; v += 256)
    Hout[(size_t)v * DIM + c] = __expf(Hcnt[(size_t)v * DIM + c] - m) * inv;
}

// ---------------------------------------------------------------------------
extern "C" void kernel_launch(void* const* d_in, const int* in_sizes, int n_in,
                              void* d_out, int out_size, void* d_ws, size_t ws_size,
                              hipStream_t stream) {
  const int*   edge_index = (const int*)  d_in[0];
  const float* features   = (const float*)d_in[1];
  const float* lin_W      = (const float*)d_in[2];
  const float* lin_b      = (const float*)d_in[3];
  const float* gcn1_W     = (const float*)d_in[4];
  const float* gcn1_b     = (const float*)d_in[5];
  const float* gcn2_W     = (const float*)d_in[6];
  const float* gcn2_b     = (const float*)d_in[7];
  const float* lin1_W     = (const float*)d_in[8];
  const float* lin1_b     = (const float*)d_in[9];

  char* ws = (char*)d_ws;
  float*    all_f = (float*)   (ws);                 // [NT,128] f32  concat(features, transformed)
  _Float16* all_h = (_Float16*)(ws +  40960000);     // [NT,128] f16
  float*    bufA  = (float*)   (ws +  61440000);     // [NT,128] f32  xw / logits scratch
  float*    bufB  = (float*)   (ws + 102400000);     // [NT,128] f32  h1 -> all_features2
  _Float16* bufH  = (_Float16*)(ws + 143360000);     // [NT,128] f16  relu'd A operand
  float*    dinv  = (float*)   (ws + 163840000);     // [NT]  deg -> rsqrt(deg)
  float*    norms = (float*)   (ws + 164160000);     // [NT]
  int*      bestn = (int*)     (ws + 164480000);     // [E]
  float*    Hcnt  = (float*)   (ws + 165120000);     // [N,128]
  int*      cls   = (int*)     (ws + 175360000);     // [NT]

  float* outH     = (float*)d_out;                          // [N,128]
  float* outHyper = outH + (size_t)N_NODES * DIM;           // [128,128]
  float* outDots  = outHyper + DIM * DIM;                   // [NT,128]

  const int B = 256;

  // 1. features -> all_feat rows [0,N) as f32 + f16
  prep_features<<<(N_NODES * DIM + B - 1) / B, B, 0, stream>>>(features, all_f, all_h, N_NODES * DIM);

  // 2. three replica linears: transformed[t] = features @ lin_W[t] + lin_b[t]
  for (int t = 0; t < T_REP; ++t) {
    gemm128_wmma<false, false><<<(N_NODES + 127) / 128, B, 0, stream>>>(
        all_h, lin_W + (size_t)t * DIM * DIM, lin_b + t * DIM,
        all_f + (size_t)(1 + t) * N_NODES * DIM,
        all_h + (size_t)(1 + t) * N_NODES * DIM, N_NODES, 1.0f);
  }

  // 3-4. row norms + cosine-sim replica argmax
  row_norms<<<(NT * 32 + B - 1) / B, B, 0, stream>>>(all_f, norms, NT);
  edge_argmax<<<(N_EDGES * 32 + B - 1) / B, B, 0, stream>>>(edge_index, all_f, norms, bestn);

  // 5. degrees (self-loop via init 1.0) -> dinv
  fill_f32<<<(NT + B - 1) / B, B, 0, stream>>>(dinv, 1.0f, NT);
  deg_count<<<(N_EDGES + 3 * N_NODES + B - 1) / B, B, 0, stream>>>(edge_index, bestn, dinv);
  make_dinv<<<(NT + B - 1) / B, B, 0, stream>>>(dinv, NT);

  // 6. GCN layer 1: x = relu(all_features)
  gemm128_wmma<true, false><<<NT / 128, B, 0, stream>>>(all_h, gcn1_W, nullptr, bufA, (_Float16*)nullptr, NT, 1.0f);
  gcn_init<<<(NT * DIM + B - 1) / B, B, 0, stream>>>(bufA, gcn1_b, dinv, bufB, NT * DIM);
  gcn_scatter<<<(TOT_E * 32 + B - 1) / B, B, 0, stream>>>(edge_index, bestn, dinv, bufA, bufB);
  relu_half<<<(NT * DIM + B - 1) / B, B, 0, stream>>>(bufB, bufH, NT * DIM);   // h1 (f16)

  // 7. GCN layer 2 -> all_features2 in bufB
  gemm128_wmma<false, false><<<NT / 128, B, 0, stream>>>(bufH, gcn2_W, nullptr, bufA, (_Float16*)nullptr, NT, 1.0f);
  gcn_init<<<(NT * DIM + B - 1) / B, B, 0, stream>>>(bufA, gcn2_b, dinv, bufB, NT * DIM);
  gcn_scatter<<<(TOT_E * 32 + B - 1) / B, B, 0, stream>>>(edge_index, bestn, dinv, bufA, bufB);

  // 8. logits = relu(all_features2) @ lin1_W + lin1_b ; per-row argmax -> classes
  relu_half<<<(NT * DIM + B - 1) / B, B, 0, stream>>>(bufB, bufH, NT * DIM);
  gemm128_wmma<false, false><<<NT / 128, B, 0, stream>>>(bufH, lin1_W, lin1_b, bufA, (_Float16*)nullptr, NT, 1.0f);
  row_argmax<<<(NT * 32 + B - 1) / B, B, 0, stream>>>(bufA, cls, NT);

  // 9. vote counts
  fill_f32<<<(N_NODES * DIM + B - 1) / B, B, 0, stream>>>(Hcnt, 0.f, N_NODES * DIM);
  vote_count<<<(NT + B - 1) / B, B, 0, stream>>>(cls, Hcnt, NT);

  // 10. hyperedge_features = mask.T @ all_features2[:N]   (directly into d_out)
  fill_f32<<<(DIM * DIM + B - 1) / B, B, 0, stream>>>(outHyper, 0.f, DIM * DIM);
  hyperedge_kernel<<<(N_NODES + HCHUNK - 1) / HCHUNK, B, 0, stream>>>(Hcnt, bufB, outHyper);

  // 11. dots = (all_features @ hyper^T) * D^-0.5  (W already [N][K] -> W_T path)
  gemm128_wmma<false, true><<<NT / 128, B, 0, stream>>>(all_h, outHyper, nullptr, outDots,
                                                        (_Float16*)nullptr, NT, 0.08838834764831845f);

  // 12. H = softmax(counts, axis=0)
  softmax_col<<<DIM, B, 0, stream>>>(Hcnt, outH);
}
